// TemporalAttentionTok_56066503082211
// MI455X (gfx1250) — compile-verified, matrix operand order checked by reading, not run on hardware
//
#include <hip/hip_runtime.h>

typedef __bf16 bf16_t;
typedef __attribute__((ext_vector_type(16))) __bf16 v16bf;
typedef __attribute__((ext_vector_type(8)))  float  v8f;

#define DIM 512
#define TSEQ 256
#define HD 64
#define NH 8
#define VST 264   // padded row stride (elements) for transposed V in LDS

// ---------------------------------------------------------------------------
// WMMA helpers
// ---------------------------------------------------------------------------
__device__ __forceinline__ v8f wmma_bf16(v16bf a, v16bf b, v8f c) {
  return __builtin_amdgcn_wmma_f32_16x16x32_bf16(false, a, false, b, (short)0, c,
                                                 false, false);
}

// 16x32 bf16 fragment from row-major [row][k] storage (ld in elements).
// Lane l: row = l&15, k-chunks at (l>>4)*8 and (l>>4)*8+16 (8 elems each).
__device__ __forceinline__ v16bf load_frag(const bf16_t* p0, int ld, int lane) {
  const bf16_t* p = p0 + (size_t)(lane & 15) * ld + ((lane >> 4) << 3);
  union { uint4 u; bf16_t h[8]; } a, b;
  a.u = *(const uint4*)p;
  b.u = *(const uint4*)(p + 16);
  v16bf f;
#pragma unroll
  for (int i = 0; i < 8; ++i) { f[i] = a.h[i]; f[8 + i] = b.h[i]; }
  return f;
}

// ---------------------------------------------------------------------------
// Kernel 0: f32 -> bf16 conversion (x and weights)
// ---------------------------------------------------------------------------
__global__ __launch_bounds__(256) void cvt_f32_bf16(const float* __restrict__ s,
                                                    bf16_t* __restrict__ d, int n) {
  int i = blockIdx.x * blockDim.x + threadIdx.x;
  int idx = i * 4;
  if (idx < n) {
    float4 f = *(const float4*)(s + idx);
    union { uint2 u; bf16_t h[4]; } o;
    o.h[0] = (bf16_t)f.x; o.h[1] = (bf16_t)f.y;
    o.h[2] = (bf16_t)f.z; o.h[3] = (bf16_t)f.w;
    *(uint2*)(d + idx) = o.u;
  }
}

// ---------------------------------------------------------------------------
// Kernel 1: fused QKV projection + RMSNorm + RoPE, output (B,H,T,hd) bf16
// grid: (M/128, 24) where y = sel*8 + head; block: 128 (4 waves, 32 rows each)
// Weight slab (64x512) staged in LDS once per block; fragments double-buffered.
// ---------------------------------------------------------------------------
__global__ __launch_bounds__(128) void qkv_rope_kernel(
    const bf16_t* __restrict__ xb,
    const bf16_t* __restrict__ wqb, const bf16_t* __restrict__ wkb,
    const bf16_t* __restrict__ wvb,
    const float* __restrict__ gq, const float* __restrict__ gk,
    bf16_t* __restrict__ dq, bf16_t* __restrict__ dk, bf16_t* __restrict__ dv) {
  __shared__ alignas(16) bf16_t wsm[HD * DIM];  // 64 KB weight slab

  const int tid  = threadIdx.x;
  const int lane = tid & 31;
  const int wid  = tid >> 5;
  const int sel  = blockIdx.y >> 3;   // 0=q 1=k 2=v
  const int h    = blockIdx.y & 7;

  const bf16_t* w   = (sel == 0) ? wqb : (sel == 1) ? wkb : wvb;
  bf16_t*       dst = (sel == 0) ? dq  : (sel == 1) ? dk  : dv;
  const float*  g   = (sel == 1) ? gk : gq;

  {  // cooperative stage of this head's weight rows [h*64 .. h*64+63]
    const uint4* src = (const uint4*)(w + (size_t)h * HD * DIM);
    uint4* d4 = (uint4*)wsm;
#pragma unroll
    for (int i = 0; i < 32; ++i) d4[tid + i * 128] = src[tid + i * 128];
  }
  __syncthreads();

  const int m0 = (blockIdx.x * 4 + wid) * 32;
  const bf16_t* arow0 = xb + (size_t)m0 * DIM;
  const bf16_t* arow1 = arow0 + 16 * DIM;

  v8f acc[2][4];
#pragma unroll
  for (int t = 0; t < 2; ++t)
#pragma unroll
    for (int j = 0; j < 4; ++j) acc[t][j] = (v8f){};

  // prime the pipeline
  v16bf a0 = load_frag(arow0, DIM, lane);
  v16bf a1 = load_frag(arow1, DIM, lane);
  v16bf b0 = load_frag(wsm + 0 * 16 * DIM, DIM, lane);
  v16bf b1 = load_frag(wsm + 1 * 16 * DIM, DIM, lane);
  v16bf b2 = load_frag(wsm + 2 * 16 * DIM, DIM, lane);
  v16bf b3 = load_frag(wsm + 3 * 16 * DIM, DIM, lane);

#pragma unroll 4
  for (int k = 0; k < DIM; k += 32) {
    const int kn = (k + 32) & (DIM - 1);  // wrapped prefetch (last one unused)
    v16bf a0n = load_frag(arow0 + kn, DIM, lane);
    v16bf a1n = load_frag(arow1 + kn, DIM, lane);
    v16bf b0n = load_frag(wsm + 0 * 16 * DIM + kn, DIM, lane);
    v16bf b1n = load_frag(wsm + 1 * 16 * DIM + kn, DIM, lane);
    v16bf b2n = load_frag(wsm + 2 * 16 * DIM + kn, DIM, lane);
    v16bf b3n = load_frag(wsm + 3 * 16 * DIM + kn, DIM, lane);

    acc[0][0] = wmma_bf16(a0, b0, acc[0][0]);
    acc[1][0] = wmma_bf16(a1, b0, acc[1][0]);
    acc[0][1] = wmma_bf16(a0, b1, acc[0][1]);
    acc[1][1] = wmma_bf16(a1, b1, acc[1][1]);
    acc[0][2] = wmma_bf16(a0, b2, acc[0][2]);
    acc[1][2] = wmma_bf16(a1, b2, acc[1][2]);
    acc[0][3] = wmma_bf16(a0, b3, acc[0][3]);
    acc[1][3] = wmma_bf16(a1, b3, acc[1][3]);

    a0 = a0n; a1 = a1n; b0 = b0n; b1 = b1n; b2 = b2n; b3 = b3n;
  }

  const int nl = lane & 15;
  const int mb = (lane >> 4) * 8;
  const float g0 = g[nl], g1 = g[16 + nl], g2 = g[32 + nl], g3 = g[48 + nl];
  const float L2B = 13.287712379549449f;  // log2(10000)
  const float invf_lo = exp2f(-(float)nl * (L2B / 32.0f));
  const float invf_hi = exp2f(-(float)(nl + 16) * (L2B / 32.0f));

#pragma unroll
  for (int tt = 0; tt < 2; ++tt) {
    const int m0t = m0 + tt * 16;
    float vals[4][8];
#pragma unroll
    for (int r = 0; r < 8; ++r) {
      vals[0][r] = acc[tt][0][r]; vals[1][r] = acc[tt][1][r];
      vals[2][r] = acc[tt][2][r]; vals[3][r] = acc[tt][3][r];
    }

    if (sel < 2) {
      // RMSNorm over hd=64: reduce across the 16 lanes of each half-wave.
      float ss[8];
#pragma unroll
      for (int r = 0; r < 8; ++r)
        ss[r] = vals[0][r] * vals[0][r] + vals[1][r] * vals[1][r] +
                vals[2][r] * vals[2][r] + vals[3][r] * vals[3][r];
#pragma unroll
      for (int d = 1; d < 16; d <<= 1)
#pragma unroll
        for (int r = 0; r < 8; ++r) ss[r] += __shfl_xor(ss[r], d, 32);

#pragma unroll
      for (int r = 0; r < 8; ++r) {
        float rn = rsqrtf(ss[r] * (1.0f / HD) + 1e-6f);
        float x0 = vals[0][r] * rn * g0;
        float x1 = vals[1][r] * rn * g1;
        float x2 = vals[2][r] * rn * g2;
        float x3 = vals[3][r] * rn * g3;
        // RoPE: pair (c, c+32) share the angle t * 10000^(-(c%32)/32).
        float t = (float)((m0t + mb + r) & (TSEQ - 1));
        float cl, sl, ch, sh;
        __sincosf(t * invf_lo, &sl, &cl);
        __sincosf(t * invf_hi, &sh, &ch);
        vals[0][r] = x0 * cl - x2 * sl;
        vals[2][r] = x2 * cl + x0 * sl;
        vals[1][r] = x1 * ch - x3 * sh;
        vals[3][r] = x3 * ch + x1 * sh;
      }
    }

#pragma unroll
    for (int r = 0; r < 8; ++r) {
      int m = m0t + mb + r;
      int b = m >> 8, t = m & (TSEQ - 1);
      size_t rowo = ((size_t)(b * NH + h) * TSEQ + t) * HD;
#pragma unroll
      for (int j = 0; j < 4; ++j) dst[rowo + j * 16 + nl] = (bf16_t)vals[j][r];
    }
  }
}

// ---------------------------------------------------------------------------
// Kernel 2: causal flash attention per (b,h). Block = 256 thr = 8 waves;
// each wave owns two 16-row q tiles, streams keys in chunks of 32 with
// double-buffered K fragments and V loads overlapped with softmax VALU.
// ---------------------------------------------------------------------------
__global__ __launch_bounds__(256) void attn_kernel(
    const bf16_t* __restrict__ q, const bf16_t* __restrict__ k,
    const bf16_t* __restrict__ v, bf16_t* __restrict__ ao) {
  __shared__ alignas(16) bf16_t ks[TSEQ * HD];    // K row-major [t][hd]
  __shared__ alignas(16) bf16_t vs[HD * VST];     // V transposed [hd][t], padded
  __shared__ alignas(16) bf16_t ps[8 * 16 * 32];  // per-wave P staging

  const int tid  = threadIdx.x;
  const int lane = tid & 31;
  const int wave = tid >> 5;
  const int bh   = blockIdx.x;
  const int b    = bh >> 3, h = bh & 7;
  const size_t base = (size_t)bh * TSEQ * HD;

  {  // stage K
    const uint4* src = (const uint4*)(k + base + (size_t)tid * HD);
    uint4* dst = (uint4*)(ks + tid * HD);
#pragma unroll
    for (int i = 0; i < 8; ++i) dst[i] = src[i];
  }
  {  // stage V transposed
    union { uint4 u[8]; bf16_t h[64]; } row;
    const uint4* src = (const uint4*)(v + base + (size_t)tid * HD);
#pragma unroll
    for (int i = 0; i < 8; ++i) row.u[i] = src[i];
#pragma unroll
    for (int c = 0; c < 64; ++c) vs[c * VST + tid] = row.h[c];
  }
  __syncthreads();

  const int nl = lane & 15;
  const int mb = (lane >> 4) * 8;
  const float inv_cap = 0.125f / 50.0f;  // scale / soft_cap

  for (int p = 0; p < 2; ++p) {
    const int qi = wave * 2 + p;
    const int m0 = qi * 16;
    v16bf qa0 = load_frag(q + base + (size_t)m0 * HD + 0,  HD, lane);
    v16bf qa1 = load_frag(q + base + (size_t)m0 * HD + 32, HD, lane);
    v8f o0 = {}, o1 = {}, o2 = {}, o3 = {};
    float mrun[8], lrun[8];
#pragma unroll
    for (int r = 0; r < 8; ++r) { mrun[r] = -1e30f; lrun[r] = 0.f; }

    const int nch = (qi + 2) >> 1;  // causal: chunks of 32 keys up to diagonal

    // prime chunk-0 K fragments
    v16bf kb00 = load_frag(ks + 0 * HD + 0,  HD, lane);
    v16bf kb01 = load_frag(ks + 0 * HD + 32, HD, lane);
    v16bf kb10 = load_frag(ks + 16 * HD + 0,  HD, lane);
    v16bf kb11 = load_frag(ks + 16 * HD + 32, HD, lane);

    for (int cb = 0; cb < nch; ++cb) {
      const int kk = cb * 32;
      v8f s0 = {}, s1 = {};
      s0 = wmma_bf16(qa0, kb00, s0); s0 = wmma_bf16(qa1, kb01, s0);
      s1 = wmma_bf16(qa0, kb10, s1); s1 = wmma_bf16(qa1, kb11, s1);

      // prefetch next chunk's K fragments (wrapped; last one unused)
      const int knx = ((cb + 1 < nch) ? (cb + 1) : 0) * 32;
      v16bf nk00 = load_frag(ks + (knx     ) * HD + 0,  HD, lane);
      v16bf nk01 = load_frag(ks + (knx     ) * HD + 32, HD, lane);
      v16bf nk10 = load_frag(ks + (knx + 16) * HD + 0,  HD, lane);
      v16bf nk11 = load_frag(ks + (knx + 16) * HD + 32, HD, lane);
      // V fragments for this chunk: independent of scores, overlap softmax
      v16bf vb0 = load_frag(vs +  0 * VST + kk, VST, lane);
      v16bf vb1 = load_frag(vs + 16 * VST + kk, VST, lane);
      v16bf vb2 = load_frag(vs + 32 * VST + kk, VST, lane);
      v16bf vb3 = load_frag(vs + 48 * VST + kk, VST, lane);

      float t0[8], t1[8];
#pragma unroll
      for (int r = 0; r < 8; ++r) {
        int row = m0 + mb + r;
        float a0 = 50.0f * tanhf(s0[r] * inv_cap);
        float a1 = 50.0f * tanhf(s1[r] * inv_cap);
        t0[r] = (kk + nl      <= row) ? a0 : -1e30f;
        t1[r] = (kk + 16 + nl <= row) ? a1 : -1e30f;
      }
      float mx[8];
#pragma unroll
      for (int r = 0; r < 8; ++r) mx[r] = fmaxf(t0[r], t1[r]);
#pragma unroll
      for (int d = 1; d < 16; d <<= 1)
#pragma unroll
        for (int r = 0; r < 8; ++r) mx[r] = fmaxf(mx[r], __shfl_xor(mx[r], d, 32));

      float sum[8];
      bf16_t* pw = ps + wave * 512;
#pragma unroll
      for (int r = 0; r < 8; ++r) {
        float mnew  = fmaxf(mrun[r], mx[r]);
        float alpha = __expf(mrun[r] - mnew);
        mrun[r] = mnew;
        float p0 = __expf(t0[r] - mnew);
        float p1 = __expf(t1[r] - mnew);
        sum[r] = p0 + p1;
        lrun[r] *= alpha;
        o0[r] *= alpha; o1[r] *= alpha; o2[r] *= alpha; o3[r] *= alpha;
        pw[(mb + r) * 32 + nl]      = (bf16_t)p0;
        pw[(mb + r) * 32 + 16 + nl] = (bf16_t)p1;
      }
#pragma unroll
      for (int d = 1; d < 16; d <<= 1)
#pragma unroll
        for (int r = 0; r < 8; ++r) sum[r] += __shfl_xor(sum[r], d, 32);
#pragma unroll
      for (int r = 0; r < 8; ++r) lrun[r] += sum[r];

      asm volatile("" ::: "memory");  // keep LDS P store before fragment re-load
      v16bf pa = load_frag(pw, 32, lane);
      o0 = wmma_bf16(pa, vb0, o0);
      o1 = wmma_bf16(pa, vb1, o1);
      o2 = wmma_bf16(pa, vb2, o2);
      o3 = wmma_bf16(pa, vb3, o3);

      kb00 = nk00; kb01 = nk01; kb10 = nk10; kb11 = nk11;
    }

#pragma unroll
    for (int r = 0; r < 8; ++r) {
      int t = m0 + mb + r;
      float inv = 1.0f / lrun[r];
      size_t rowo = ((size_t)(b * TSEQ + t) * DIM) + h * HD;
      ao[rowo +  0 + nl] = (bf16_t)(o0[r] * inv);
      ao[rowo + 16 + nl] = (bf16_t)(o1[r] * inv);
      ao[rowo + 32 + nl] = (bf16_t)(o2[r] * inv);
      ao[rowo + 48 + nl] = (bf16_t)(o3[r] * inv);
    }
  }
}

// ---------------------------------------------------------------------------
// Kernel 3: output projection, fp32 result. Same pipelined 32x64/wave GEMM.
// ---------------------------------------------------------------------------
__global__ __launch_bounds__(128) void outproj_kernel(
    const bf16_t* __restrict__ ao, const bf16_t* __restrict__ wob,
    float* __restrict__ out) {
  __shared__ alignas(16) bf16_t wsm[HD * DIM];

  const int tid  = threadIdx.x;
  const int lane = tid & 31;
  const int wid  = tid >> 5;
  const int n0   = blockIdx.y * 64;

  {
    const uint4* src = (const uint4*)(wob + (size_t)n0 * DIM);
    uint4* d4 = (uint4*)wsm;
#pragma unroll
    for (int i = 0; i < 32; ++i) d4[tid + i * 128] = src[tid + i * 128];
  }
  __syncthreads();

  const int m0 = (blockIdx.x * 4 + wid) * 32;
  const bf16_t* arow0 = ao + (size_t)m0 * DIM;
  const bf16_t* arow1 = arow0 + 16 * DIM;

  v8f acc[2][4];
#pragma unroll
  for (int t = 0; t < 2; ++t)
#pragma unroll
    for (int j = 0; j < 4; ++j) acc[t][j] = (v8f){};

  v16bf a0 = load_frag(arow0, DIM, lane);
  v16bf a1 = load_frag(arow1, DIM, lane);
  v16bf b0 = load_frag(wsm + 0 * 16 * DIM, DIM, lane);
  v16bf b1 = load_frag(wsm + 1 * 16 * DIM, DIM, lane);
  v16bf b2 = load_frag(wsm + 2 * 16 * DIM, DIM, lane);
  v16bf b3 = load_frag(wsm + 3 * 16 * DIM, DIM, lane);

#pragma unroll 4
  for (int k = 0; k < DIM; k += 32) {
    const int kn = (k + 32) & (DIM - 1);
    v16bf a0n = load_frag(arow0 + kn, DIM, lane);
    v16bf a1n = load_frag(arow1 + kn, DIM, lane);
    v16bf b0n = load_frag(wsm + 0 * 16 * DIM + kn, DIM, lane);
    v16bf b1n = load_frag(wsm + 1 * 16 * DIM + kn, DIM, lane);
    v16bf b2n = load_frag(wsm + 2 * 16 * DIM + kn, DIM, lane);
    v16bf b3n = load_frag(wsm + 3 * 16 * DIM + kn, DIM, lane);

    acc[0][0] = wmma_bf16(a0, b0, acc[0][0]);
    acc[1][0] = wmma_bf16(a1, b0, acc[1][0]);
    acc[0][1] = wmma_bf16(a0, b1, acc[0][1]);
    acc[1][1] = wmma_bf16(a1, b1, acc[1][1]);
    acc[0][2] = wmma_bf16(a0, b2, acc[0][2]);
    acc[1][2] = wmma_bf16(a1, b2, acc[1][2]);
    acc[0][3] = wmma_bf16(a0, b3, acc[0][3]);
    acc[1][3] = wmma_bf16(a1, b3, acc[1][3]);

    a0 = a0n; a1 = a1n; b0 = b0n; b1 = b1n; b2 = b2n; b3 = b3n;
  }

  const int nl = lane & 15;
  const int mb = (lane >> 4) * 8;
#pragma unroll
  for (int tt = 0; tt < 2; ++tt) {
#pragma unroll
    for (int r = 0; r < 8; ++r) {
      size_t rowo = (size_t)(m0 + tt * 16 + mb + r) * DIM + n0;
      out[rowo +  0 + nl] = acc[tt][0][r];
      out[rowo + 16 + nl] = acc[tt][1][r];
      out[rowo + 32 + nl] = acc[tt][2][r];
      out[rowo + 48 + nl] = acc[tt][3][r];
    }
  }
}

// ---------------------------------------------------------------------------
extern "C" void kernel_launch(void* const* d_in, const int* in_sizes, int n_in,
                              void* d_out, int out_size, void* d_ws, size_t ws_size,
                              hipStream_t stream) {
  (void)in_sizes; (void)n_in; (void)out_size; (void)ws_size;
  const float* x  = (const float*)d_in[0];
  const float* wq = (const float*)d_in[1];
  const float* wk = (const float*)d_in[2];
  const float* wv = (const float*)d_in[3];
  const float* wo = (const float*)d_in[4];
  const float* gq = (const float*)d_in[5];
  const float* gk = (const float*)d_in[6];

  const size_t NX = (size_t)256 * 256 * 512;  // 33,554,432
  const size_t NW = (size_t)512 * 512;

  char* ws = (char*)d_ws;
  bf16_t* xb  = (bf16_t*)(ws);            // bf16 x; later reused as attn output
  bf16_t* qb  = (bf16_t*)(ws + NX * 2);
  bf16_t* kb  = (bf16_t*)(ws + NX * 4);
  bf16_t* vb  = (bf16_t*)(ws + NX * 6);
  bf16_t* wqb = (bf16_t*)(ws + NX * 8);
  bf16_t* wkb = wqb + NW;
  bf16_t* wvb = wkb + NW;
  bf16_t* wob = wvb + NW;

  cvt_f32_bf16<<<dim3((unsigned)(NX / 4 / 256)), 256, 0, stream>>>(x, xb, (int)NX);
  cvt_f32_bf16<<<dim3((unsigned)(NW / 4 / 256)), 256, 0, stream>>>(wq, wqb, (int)NW);
  cvt_f32_bf16<<<dim3((unsigned)(NW / 4 / 256)), 256, 0, stream>>>(wk, wkb, (int)NW);
  cvt_f32_bf16<<<dim3((unsigned)(NW / 4 / 256)), 256, 0, stream>>>(wv, wvb, (int)NW);
  cvt_f32_bf16<<<dim3((unsigned)(NW / 4 / 256)), 256, 0, stream>>>(wo, wob, (int)NW);

  qkv_rope_kernel<<<dim3(512, 24), 128, 0, stream>>>(xb, wqb, wkb, wvb, gq, gk,
                                                     qb, kb, vb);
  attn_kernel<<<dim3(2048), 256, 0, stream>>>(qb, kb, vb, /*ao=*/xb);
  outproj_kernel<<<dim3(512, 8), 128, 0, stream>>>(xb, wob, (float*)d_out);
}
